// FFTDiffuseBase_85607288143898
// MI455X (gfx1250) — matrix operand order repaired and checked by reading, not compiled, not run on hardware
//
#include <hip/hip_runtime.h>

// ---------------------------------------------------------------------------
// FFT-guided diffusion for MI455X (gfx1250, wave32, WMMA).
//
//  - per-block spectral operator op^10 == 10 periodic constant-coefficient
//    diffusion steps  B <- Ay*B + B*Bx  (circulant tridiagonal Ay, Bx)
//    -> dense 128x128 matmuls on the WMMA pipes (v_wmma_f32_16x16x32_f16,
//    f32 accumulate), block + transposed copy resident in 192 KB LDS so all
//    fragment loads are contiguous ds_load_b128 (no column gathers).
//  - 250 explicit stencil iterations: L2-resident ping-pong VALU kernels.
// ---------------------------------------------------------------------------

#define HH 512
#define WW 512
#define BATCH 2
#define BSZ 128
#define OVL 16
#define STRIDE 112
#define NPOS 5
#define NITER 250
#define LCOEF 0.24f
#define KCOND 0.03f
#define FFT_STEPS 10
#define DEPS 0.1f
#define THRU 0.1f

typedef __attribute__((ext_vector_type(16))) _Float16 v16h;
typedef __attribute__((ext_vector_type(8)))  _Float16 v8h;
typedef __attribute__((ext_vector_type(8)))  float    v8f;

__device__ __forceinline__ int imin(int a, int b) { return a < b ? a : b; }

// ---------------------------------------------------------------------------
// shift = DEPS if min(initial) <= DEPS else 0   (single-workgroup reduction)
// ---------------------------------------------------------------------------
__global__ void min_shift_kernel(const float* __restrict__ initial,
                                 float* __restrict__ shift, int n) {
  __shared__ float red[1024];
  float m = 3.4e38f;
  for (int i = threadIdx.x; i < n; i += blockDim.x) m = fminf(m, initial[i]);
  red[threadIdx.x] = m;
  __syncthreads();
  for (int s = blockDim.x >> 1; s > 0; s >>= 1) {
    if ((int)threadIdx.x < s) red[threadIdx.x] = fminf(red[threadIdx.x], red[threadIdx.x + s]);
    __syncthreads();
  }
  if (threadIdx.x == 0) shift[0] = (red[0] <= DEPS) ? DEPS : 0.0f;
}

__global__ void add_shift_kernel(const float* __restrict__ in,
                                 const float* __restrict__ shift,
                                 float* __restrict__ out, int n) {
  int i = blockIdx.x * blockDim.x + threadIdx.x;
  if (i < n) out[i] = in[i] + shift[0];
}

// ---------------------------------------------------------------------------
// cv[b, H-1, W], ch[b, H, W-1] : g(mean_c |grad guide|, K)
// ---------------------------------------------------------------------------
__global__ void conductance_kernel(const float* __restrict__ guide,
                                   float* __restrict__ cv,
                                   float* __restrict__ ch) {
  int idx = blockIdx.x * blockDim.x + threadIdx.x;
  if (idx >= BATCH * HH * WW) return;
  int b = idx / (HH * WW);
  int r = idx % (HH * WW);
  int y = r / WW, x = r % WW;
  const float* g = guide + (size_t)b * 3 * HH * WW;
  const float invK2 = 1.0f / (KCOND * KCOND);
  if (y < HH - 1) {
    float s = 0.f;
    for (int c = 0; c < 3; c++)
      s += fabsf(g[c * HH * WW + (y + 1) * WW + x] - g[c * HH * WW + y * WW + x]);
    float m = s * (1.f / 3.f);
    cv[(size_t)b * (HH - 1) * WW + y * WW + x] = 1.0f / (1.0f + m * m * invK2);
  }
  if (x < WW - 1) {
    float s = 0.f;
    for (int c = 0; c < 3; c++)
      s += fabsf(g[c * HH * WW + y * WW + x + 1] - g[c * HH * WW + y * WW + x]);
    float m = s * (1.f / 3.f);
    ch[(size_t)b * HH * (WW - 1) + y * (WW - 1) + x] = 1.0f / (1.0f + m * m * invK2);
  }
}

// ---------------------------------------------------------------------------
// ur[b, H-1, W-1] = (var3(cv) < thr) & (var3(ch) < thr), zero-padded 3x3 box
// ---------------------------------------------------------------------------
__global__ void uniform_kernel(const float* __restrict__ cv,
                               const float* __restrict__ ch,
                               float* __restrict__ ur) {
  int idx = blockIdx.x * blockDim.x + threadIdx.x;
  const int HM = HH - 1, WM = WW - 1;
  if (idx >= BATCH * HM * WM) return;
  int b = idx / (HM * WM);
  int r = idx % (HM * WM);
  int y = r / WM, x = r % WM;

  const float* CV = cv + (size_t)b * (HH - 1) * WW;     // (H-1) x W
  const float* CH = ch + (size_t)b * HH * (WW - 1);     // H x (W-1)
  float s1 = 0.f, s2 = 0.f, t1 = 0.f, t2 = 0.f;
  for (int dy = -1; dy <= 1; dy++)
    for (int dx = -1; dx <= 1; dx++) {
      int yy = y + dy, xx = x + dx;
      if (yy >= 0 && yy < HH - 1 && xx >= 0 && xx < WW) {
        float v = CV[yy * WW + xx];
        s1 += v; s2 += v * v;
      }
      if (yy >= 0 && yy < HH && xx >= 0 && xx < WW - 1) {
        float v = CH[yy * (WW - 1) + xx];
        t1 += v; t2 += v * v;
      }
    }
  float cvv = s2 * (1.f / 9.f) - (s1 * (1.f / 9.f)) * (s1 * (1.f / 9.f));
  float chv = t2 * (1.f / 9.f) - (t1 * (1.f / 9.f)) * (t1 * (1.f / 9.f));
  ur[idx] = (cvv < THRU && chv < THRU) ? 1.0f : 0.0f;
}

// ---------------------------------------------------------------------------
// Per-block means of cv, ch, ur  ->  stats[(b*25 + yi*5 + xi)*3 + {0,1,2}]
// ---------------------------------------------------------------------------
__global__ void block_stats_kernel(const float* __restrict__ cv,
                                   const float* __restrict__ ch,
                                   const float* __restrict__ ur,
                                   float* __restrict__ stats) {
  __shared__ float r0[256], r1[256], r2[256];
  int xi = blockIdx.x, yi = blockIdx.y, b = blockIdx.z;
  int y0 = yi * STRIDE, x0 = xi * STRIDE;
  int y_end = imin(y0 + BSZ, HH), x_end = imin(x0 + BSZ, WW);
  int cv_ye = imin(y_end - 1, HH - 1), cv_xe = x_end;
  int ch_ye = y_end, ch_xe = imin(x_end - 1, WW - 1);
  int ur_ye = imin(y_end, HH - 1), ur_xe = imin(x_end, WW - 1);

  const float* CV = cv + (size_t)b * (HH - 1) * WW;
  const float* CH = ch + (size_t)b * HH * (WW - 1);
  const float* UR = ur + (size_t)b * (HH - 1) * (WW - 1);

  float scv = 0.f, sch = 0.f, sur = 0.f;
  int rh, rw;
  rh = cv_ye - y0; rw = cv_xe - x0;
  for (int t = threadIdx.x; t < rh * rw; t += blockDim.x)
    scv += CV[(y0 + t / rw) * WW + x0 + t % rw];
  float ncv = (float)(rh * rw);
  rh = ch_ye - y0; rw = ch_xe - x0;
  for (int t = threadIdx.x; t < rh * rw; t += blockDim.x)
    sch += CH[(y0 + t / rw) * (WW - 1) + x0 + t % rw];
  float nch = (float)(rh * rw);
  rh = ur_ye - y0; rw = ur_xe - x0;
  for (int t = threadIdx.x; t < rh * rw; t += blockDim.x)
    sur += UR[(y0 + t / rw) * (WW - 1) + x0 + t % rw];
  float nur = (float)(rh * rw);

  r0[threadIdx.x] = scv; r1[threadIdx.x] = sch; r2[threadIdx.x] = sur;
  __syncthreads();
  for (int s = blockDim.x >> 1; s > 0; s >>= 1) {
    if ((int)threadIdx.x < s) {
      r0[threadIdx.x] += r0[threadIdx.x + s];
      r1[threadIdx.x] += r1[threadIdx.x + s];
      r2[threadIdx.x] += r2[threadIdx.x + s];
    }
    __syncthreads();
  }
  if (threadIdx.x == 0) {
    float* st = stats + ((size_t)b * NPOS * NPOS + yi * NPOS + xi) * 3;
    st[0] = r0[0] / ncv;
    st[1] = r1[0] / nch;
    st[2] = r2[0] / nur;
  }
}

// ---------------------------------------------------------------------------
// WMMA fragment loaders per CDNA5 ISA layout tables (wave32).
// All loads below are contiguous per lane -> lower to ds_load_b128 pairs.
// ---------------------------------------------------------------------------
// 16-bit A 16x32 (row-major source): lanes 0-15 -> M=lane,
//   K={k0..k0+7, k0+16..k0+23}; lanes 16-31 -> M=lane-16, K shifted by 8.
__device__ __forceinline__ v16h load_a_frag(const _Float16* M, int ld,
                                            int r0, int k0, int lane) {
  v16h a;
  int m = r0 + (lane & 15);
  int kh = (lane >> 4) * 8;
  const _Float16* row = M + m * ld + k0;
#pragma unroll
  for (int j = 0; j < 8; j++) a[j] = row[kh + j];
#pragma unroll
  for (int j = 0; j < 8; j++) a[8 + j] = row[16 + kh + j];
  return a;
}

// 16-bit B 32x16 fragment from a TRANSPOSED (or symmetric) source Mt where
// Mt[n*ld + k] == B[k][n]:  lanes 0-15 -> N=lane, K={k0..k0+7, k0+16..k0+23};
// lanes 16-31 -> N=lane-16, K shifted by 8 (matches load_a_frag).
__device__ __forceinline__ v16h load_b_frag_tr(const _Float16* Mt, int ld,
                                               int k0, int c0, int lane) {
  v16h b;
  int n = c0 + (lane & 15);
  int kh = (lane >> 4) * 8;
  const _Float16* base = Mt + n * ld + k0;
#pragma unroll
  for (int j = 0; j < 8; j++) b[j] = base[kh + j];
#pragma unroll
  for (int j = 0; j < 8; j++) b[8 + j] = base[16 + kh + j];
  return b;
}

// ---------------------------------------------------------------------------
// Block spectral diffusion: 10x (cur <- Ay*cur + cur*Bx), periodic circulant
// tridiagonal Ay (h x h, includes identity) and Bx (w x w). This is exactly
// irfft2(rfft2(block) * op) of the reference. Gated on unif > 0.7, blended
// with the 16-pixel overlap ramps at writeback. Dynamic LDS: 192 KB.
// cur is kept in row-major (A operand) and transposed (B operand) layouts.
// ---------------------------------------------------------------------------
__global__ void block_diffuse_kernel(float* __restrict__ depth,
                                     const float* __restrict__ stats,
                                     int y0, int x0) {
  extern __shared__ char smem[];
  int b = blockIdx.x;
  int h = imin(y0 + BSZ, HH) - y0;
  int w = imin(x0 + BSZ, WW) - x0;
  int yi = y0 / STRIDE, xi = x0 / STRIDE;
  const float* st = stats + ((size_t)b * NPOS * NPOS + yi * NPOS + xi) * 3;
  float cvm = st[0], chm = st[1], unif = st[2];
  if (unif <= 0.7f) return;  // block unchanged (upd = old)

  _Float16* Ay   = (_Float16*)smem;           // h x h  row-major (symmetric)
  _Float16* Bx   = Ay + BSZ * BSZ;            // w x w  row-major (symmetric)
  _Float16* buf0  = Bx + BSZ * BSZ;           // cur  (h x w row-major)
  _Float16* buf0T = buf0 + BSZ * BSZ;         // curT (w x h)
  _Float16* buf1  = buf0T + BSZ * BSZ;        // nxt
  _Float16* buf1T = buf1 + BSZ * BSZ;         // nxtT

  // Ay = I + L*cvm*(Sy + Sy^T - 2I)  (periodic), Bx = L*chm*(Sx + Sx^T - 2I)
  for (int t = threadIdx.x; t < h * h; t += blockDim.x) {
    int i = t / h, k = t % h;
    float v = 0.f;
    if (k == i) v = 1.f - 2.f * LCOEF * cvm;
    else if (k == (i + 1) % h || k == (i + h - 1) % h) v = LCOEF * cvm;
    Ay[t] = (_Float16)v;
  }
  for (int t = threadIdx.x; t < w * w; t += blockDim.x) {
    int k = t / w, j = t % w;
    float v = 0.f;
    if (k == j) v = -2.f * LCOEF * chm;
    else if (k == (j + 1) % w || k == (j + w - 1) % w) v = LCOEF * chm;
    Bx[t] = (_Float16)v;
  }

  float* dimg = depth + (size_t)b * HH * WW;
  for (int t = threadIdx.x; t < h * w; t += blockDim.x) {
    int i = t / w, j = t % w;
    _Float16 v = (_Float16)dimg[(y0 + i) * WW + x0 + j];
    buf0[t] = v;
    buf0T[j * h + i] = v;
  }
  __syncthreads();

  _Float16* cur  = buf0;
  _Float16* curT = buf0T;
  _Float16* nxt  = buf1;
  _Float16* nxtT = buf1T;
  int lane = threadIdx.x & 31;
  int wave = threadIdx.x >> 5;
  int nwaves = blockDim.x >> 5;  // 16 waves
  int ntx = w >> 4, nty = h >> 4, ntiles = nty * ntx;  // up to 64 tiles

  for (int s = 0; s < FFT_STEPS; s++) {
    v8f accs[4] = {};
#pragma unroll
    for (int q = 0; q < 4; q++) {
      int t = wave + q * nwaves;           // wave-uniform predicate: EXEC stays
      if (t < ntiles) {                    // all-ones inside the WMMA region
        int ti = t / ntx, tj = t % ntx;
        v8f acc = {};
        for (int kt = 0; kt < (h >> 5); kt++) {   // Ay(h x h) * cur(h x w)
          v16h a  = load_a_frag(Ay, h, ti * 16, kt * 32, lane);
          v16h bb = load_b_frag_tr(curT, h, kt * 32, tj * 16, lane);
          acc = __builtin_amdgcn_wmma_f32_16x16x32_f16(false, a, false, bb,
                                                       (short)0, acc, false, false);
        }
        for (int kt = 0; kt < (w >> 5); kt++) {   // cur(h x w) * Bx(w x w)
          v16h a  = load_a_frag(cur, w, ti * 16, kt * 32, lane);
          // Bx symmetric: its rows already give B[k][n] contiguous in k
          v16h bb = load_b_frag_tr(Bx, w, kt * 32, tj * 16, lane);
          acc = __builtin_amdgcn_wmma_f32_16x16x32_f16(false, a, false, bb,
                                                       (short)0, acc, false, false);
        }
        accs[q] = acc;
      }
    }
    __syncthreads();
#pragma unroll
    for (int q = 0; q < 4; q++) {
      int t = wave + q * nwaves;
      if (t < ntiles) {
        int ti = t / ntx, tj = t % ntx;
        int rbase = ti * 16 + ((lane >= 16) ? 8 : 0);
        int col = tj * 16 + (lane & 15);
        v8h pk;
#pragma unroll
        for (int j = 0; j < 8; j++) {
          _Float16 v = (_Float16)accs[q][j];
          pk[j] = v;
          nxt[(rbase + j) * w + col] = v;        // row-major copy (scattered)
        }
        *(v8h*)(nxtT + col * h + rbase) = pk;    // transposed copy (1x b128)
      }
    }
    __syncthreads();
    _Float16* tp;
    tp = cur;  cur  = nxt;  nxt  = tp;
    tp = curT; curT = nxtT; nxtT = tp;
  }

  // writeback with overlap blend (old still lives in global memory)
  for (int t = threadIdx.x; t < h * w; t += blockDim.x) {
    int i = t / w, j = t % w;
    int gy = y0 + i, gx = x0 + j;
    float oldv = dimg[gy * WW + gx];
    float newv = (float)cur[i * w + j];
    float by = (y0 > 0 && i < OVL) ? (float)i * (1.f / OVL) : 1.f;
    float bx = (x0 > 0 && j < OVL) ? (float)j * (1.f / OVL) : 1.f;
    float bl = by * bx;
    dimg[gy * WW + gx] = oldv * (1.f - bl) + newv * bl;
  }
}

// ---------------------------------------------------------------------------
// One explicit diffusion iteration (ping-pong); L2-resident (~8 MB set).
// ---------------------------------------------------------------------------
__global__ void diffuse_step_kernel(const float* __restrict__ cur,
                                    const float* __restrict__ cv,
                                    const float* __restrict__ ch,
                                    float* __restrict__ nxt) {
  int idx = blockIdx.x * blockDim.x + threadIdx.x;
  if (idx >= BATCH * HH * WW) return;
  int b = idx / (HH * WW);
  int r = idx % (HH * WW);
  int y = r / WW, x = r % WW;
  const float* I  = cur + (size_t)b * HH * WW;
  const float* CV = cv + (size_t)b * (HH - 1) * WW;
  const float* CH = ch + (size_t)b * HH * (WW - 1);
  float c = I[y * WW + x];
  float acc = c;
  if (y < HH - 1) acc += LCOEF * CV[y * WW + x] * (I[(y + 1) * WW + x] - c);
  if (y > 0)      acc -= LCOEF * CV[(y - 1) * WW + x] * (c - I[(y - 1) * WW + x]);
  if (x < WW - 1) acc += LCOEF * CH[y * (WW - 1) + x] * (I[y * WW + x + 1] - c);
  if (x > 0)      acc -= LCOEF * CH[y * (WW - 1) + x - 1] * (c - I[y * WW + x - 1]);
  nxt[idx] = acc;
}

__global__ void final_kernel(const float* __restrict__ cur,
                             const float* __restrict__ shift,
                             float* __restrict__ out, int n) {
  int i = blockIdx.x * blockDim.x + threadIdx.x;
  if (i < n) out[i] = cur[i] - shift[0];
}

// ---------------------------------------------------------------------------
extern "C" void kernel_launch(void* const* d_in, const int* in_sizes, int n_in,
                              void* d_out, int out_size, void* d_ws, size_t ws_size,
                              hipStream_t stream) {
  (void)in_sizes; (void)n_in; (void)out_size; (void)ws_size;
  const float* guide   = (const float*)d_in[0];
  const float* initial = (const float*)d_in[1];

  // workspace layout (floats), ~10.5 MB total
  const size_t nImg = (size_t)BATCH * HH * WW;
  float* ws    = (float*)d_ws;
  float* bufA  = ws;
  float* bufB  = bufA + nImg;
  float* cv    = bufB + nImg;
  float* ch    = cv + (size_t)BATCH * (HH - 1) * WW;
  float* ur    = ch + (size_t)BATCH * HH * (WW - 1);
  float* stats = ur + (size_t)BATCH * (HH - 1) * (WW - 1);
  float* shift = stats + BATCH * NPOS * NPOS * 3;

  const int threads = 256;
  const int blocksImg = (int)((nImg + threads - 1) / threads);

  min_shift_kernel<<<1, 1024, 0, stream>>>(initial, shift, (int)nImg);
  add_shift_kernel<<<blocksImg, threads, 0, stream>>>(initial, shift, bufA, (int)nImg);
  conductance_kernel<<<blocksImg, threads, 0, stream>>>(guide, cv, ch);

  const int nUr = BATCH * (HH - 1) * (WW - 1);
  uniform_kernel<<<(nUr + threads - 1) / threads, threads, 0, stream>>>(cv, ch, ur);
  block_stats_kernel<<<dim3(NPOS, NPOS, BATCH), 256, 0, stream>>>(cv, ch, ur, stats);

  // 25 sequential block positions (preserves reference overlap ordering);
  // the two batches are disjoint images -> parallel in gridDim.x.
  for (int yi = 0; yi < NPOS; yi++)
    for (int xi = 0; xi < NPOS; xi++)
      block_diffuse_kernel<<<BATCH, 512, 192 * 1024, stream>>>(
          bufA, stats, yi * STRIDE, xi * STRIDE);

  float* cur = bufA;
  float* nxt = bufB;
  for (int it = 0; it < NITER; it++) {
    diffuse_step_kernel<<<blocksImg, threads, 0, stream>>>(cur, cv, ch, nxt);
    float* t = cur; cur = nxt; nxt = t;
  }
  final_kernel<<<blocksImg, threads, 0, stream>>>(cur, shift, (float*)d_out, (int)nImg);
}
